// PointNetSetAbstraction_88270167868036
// MI455X (gfx1250) — matrix operand order repaired and checked by reading, not compiled, not run on hardware
//
#include <hip/hip_runtime.h>
#include <hip/hip_bf16.h>

typedef _Float16 f16;
typedef f16   v16h __attribute__((ext_vector_type(16)));
typedef f16   v8h  __attribute__((ext_vector_type(8)));
typedef float v8f  __attribute__((ext_vector_type(8)));

#define NB     8
#define NPTS   16384
#define NPOINT 512
#define NSAMP  64
#define R2     0.25f
#define ROWS   (NB * NPOINT * NSAMP)   /* 262144 flattened (b,s,k) rows */
#define C0     64
#define C1     128

union Frag16 { v16h v; v8h h[2]; };

// ---------------------------------------------------------------- FPS
__global__ __launch_bounds__(1024)
void fps_kernel(const float* __restrict__ xyz, int* __restrict__ fps_idx,
                float* __restrict__ new_xyz /* d_out head */) {
    const int b = blockIdx.x, tid = threadIdx.x;
    const int PT = NPTS / 1024;  // 16 points per thread, kept in registers
    float px[16], py[16], pz[16], dist[16];
#pragma unroll
    for (int j = 0; j < PT; ++j) {
        int i = j * 1024 + tid;
        const float* p = xyz + ((size_t)b * NPTS + i) * 3;
        px[j] = p[0]; py[j] = p[1]; pz[j] = p[2]; dist[j] = 1e10f;
    }
    __shared__ float sval[1024];
    __shared__ int   sidx[1024];
    __shared__ float sc[3];
    __shared__ int   sgi;
    if (tid == 0) sgi = 0;
    __syncthreads();
    for (int it = 0; it < NPOINT; ++it) {
        if (tid == 0) {
            int gi = sgi;
            fps_idx[b * NPOINT + it] = gi;
            const float* p = xyz + ((size_t)b * NPTS + gi) * 3;
            sc[0] = p[0]; sc[1] = p[1]; sc[2] = p[2];
            float* o = new_xyz + ((size_t)b * NPOINT + it) * 3;
            o[0] = p[0]; o[1] = p[1]; o[2] = p[2];
        }
        __syncthreads();
        float cx = sc[0], cy = sc[1], cz = sc[2];
        float best = -1.0f; int bi = 0;
#pragma unroll
        for (int j = 0; j < PT; ++j) {
            float dx = px[j] - cx, dy = py[j] - cy, dz = pz[j] - cz;
            float d = dx * dx + dy * dy + dz * dz;
            dist[j] = fminf(dist[j], d);
            if (dist[j] > best) { best = dist[j]; bi = j * 1024 + tid; }
        }
        sval[tid] = best; sidx[tid] = bi;
        __syncthreads();
        for (int off = 512; off > 0; off >>= 1) {
            if (tid < off) {
                float v2 = sval[tid + off]; int i2 = sidx[tid + off];
                if (v2 > sval[tid] || (v2 == sval[tid] && i2 < sidx[tid])) {
                    sval[tid] = v2; sidx[tid] = i2;
                }
            }
            __syncthreads();
        }
        if (tid == 0) sgi = sidx[0];
        __syncthreads();
    }
}

// ------------------------------------------------- W1 (f32->f16, [n][k])
__global__ void convert_w1(const float* __restrict__ W1, f16* __restrict__ W1h) {
    int i = blockIdx.x * 256 + threadIdx.x;
    if (i < C1 * C0) W1h[i] = (f16)W1[i];
}

// ---------------------- ball query (ordered wave32 compaction) + GEMM0
__global__ __launch_bounds__(32)
void ballquery_gemm0(const float* __restrict__ xyz, const float* __restrict__ pts,
                     const int* __restrict__ fps_idx,
                     const float* __restrict__ W0, const float* __restrict__ b0,
                     f16* __restrict__ h0h) {
    const int bs = blockIdx.x;          // (b,s)
    const int b = bs >> 9;
    const int lane = threadIdx.x;       // single wave32
    __shared__ int   sel[NSAMP];
    __shared__ float sW0[C0 * 6];
    __shared__ float sb0[C0];
    __shared__ float c3[3];
    for (int i = lane; i < C0 * 6; i += 32) sW0[i] = W0[i];
    for (int i = lane; i < C0; i += 32) sb0[i] = b0[i];
    if (lane == 0) {
        int gi = fps_idx[bs];
        const float* p = xyz + ((size_t)b * NPTS + gi) * 3;
        c3[0] = p[0]; c3[1] = p[1]; c3[2] = p[2];
    }
    __syncthreads();
    const float cx = c3[0], cy = c3[1], cz = c3[2];
    int count = 0;
    for (int base = 0; base < NPTS && count < NSAMP; base += 32) {
        int i = base + lane;
        const float* p = xyz + ((size_t)b * NPTS + i) * 3;
        float dx = p[0] - cx, dy = p[1] - cy, dz = p[2] - cz;
        bool ok = (dx * dx + dy * dy + dz * dz) <= R2;
        unsigned m = (unsigned)__ballot(ok);
        int pre = __popc(m & ((1u << lane) - 1u));
        if (ok) {
            int pos = count + pre;
            if (pos < NSAMP) sel[pos] = i;
        }
        count += __popc(m);
    }
    __syncthreads();
    int cnt = count < NSAMP ? count : NSAMP;
    int sel0 = sel[0];                  // first in-radius index (pad value)
    for (int p = cnt + lane; p < NSAMP; p += 32) sel[p] = sel0;
    __syncthreads();
    // tiny GEMM0: 6 -> 64 channels, store pre-BN f16
    for (int k = lane; k < NSAMP; k += 32) {
        int idx = sel[k];
        const float* pc = xyz + ((size_t)b * NPTS + idx) * 3;
        const float* pf = pts + ((size_t)b * NPTS + idx) * 3;
        float f0 = pc[0] - cx, f1 = pc[1] - cy, f2 = pc[2] - cz;
        float f3 = pf[0], f4 = pf[1], f5 = pf[2];
        size_t row = (size_t)bs * NSAMP + k;
#pragma unroll 4
        for (int o = 0; o < C0; ++o) {
            const float* w = sW0 + o * 6;
            float acc = sb0[o] + w[0] * f0 + w[1] * f1 + w[2] * f2
                                + w[3] * f3 + w[4] * f4 + w[5] * f5;
            h0h[row * C0 + o] = (f16)acc;
        }
    }
}

// ------------------- per-channel sum/sumsq partials (deterministic tree)
__global__ __launch_bounds__(256)
void stats_partial(const f16* __restrict__ buf, float* __restrict__ partial,
                   size_t total, int cmask, int C) {
    const int tid = threadIdx.x;
    const size_t stride = (size_t)gridDim.x * 256;   // multiple of C
    float s = 0.f, q = 0.f;
    for (size_t i = (size_t)blockIdx.x * 256 + tid; i < total; i += stride) {
        float v = (float)buf[i];
        s += v; q += v * v;
    }
    __shared__ float ss[256], sq[256];
    ss[tid] = s; sq[tid] = q;
    __syncthreads();
    for (int off = 128; off >= C; off >>= 1) {
        if (tid < off) { ss[tid] += ss[tid + off]; sq[tid] += sq[tid + off]; }
        __syncthreads();
    }
    if (tid < C) {
        partial[(size_t)blockIdx.x * 2 * C + tid]     = ss[tid];
        partial[(size_t)blockIdx.x * 2 * C + C + tid] = sq[tid];
    }
}

__global__ void finalize_bn(const float* __restrict__ partial, int nblocks, int C,
                            float count, const float* __restrict__ g,
                            const float* __restrict__ beta, float* __restrict__ bn) {
    int c = threadIdx.x;
    if (c >= C) return;
    float s = 0.f, q = 0.f;
    for (int blk = 0; blk < nblocks; ++blk) {
        s += partial[(size_t)blk * 2 * C + c];
        q += partial[(size_t)blk * 2 * C + C + c];
    }
    float mu = s / count;
    float var = q / count - mu * mu;
    float sc = g[c] * rsqrtf(var + 1e-5f);
    bn[c]     = sc;
    bn[C + c] = beta[c] - mu * sc;
}

// --------------------------- GEMM1 (64 -> 128) with v_wmma_f32_16x16x32_f16
__global__ __launch_bounds__(256)
void gemm1_wmma(const f16* __restrict__ h0h, const f16* __restrict__ W1h,
                const float* __restrict__ bn0, const float* __restrict__ b1,
                f16* __restrict__ h1h) {
    __shared__ f16 As[16 * C0];         // 16 rows x K=64, row-major
    const int tid = threadIdx.x;
    const size_t M0 = (size_t)blockIdx.x * 16;
    const float* scale0 = bn0;
    const float* shift0 = bn0 + C0;
    // stage A tile with BN0 + ReLU applied, f32 -> f16
    for (int i = tid; i < 16 * C0; i += 256) {
        int c = i & (C0 - 1);
        float v = (float)h0h[M0 * C0 + i];
        As[i] = (f16)fmaxf(scale0[c] * v + shift0[c], 0.0f);
    }
    __syncthreads();
    const int wave = tid >> 5, lane = tid & 31;
    const int n  = lane & 15;
    const int hh = lane >> 4;
    const int n0 = wave * 16;
    const int m  = lane & 15;
    // B fragments: lane holds W1h[n0+n][hh*16 .. +15] (contiguous 32B)
    Frag16 fb0, fb1;
    fb0.v = *(const v16h*)(W1h + (size_t)(n0 + n) * C0 + hh * 16);
    fb1.v = *(const v16h*)(W1h + (size_t)(n0 + n) * C0 + 32 + hh * 16);
    // A fragments: two contiguous 16B chunks per lane per K-step
    Frag16 fa0, fa1;
    fa0.h[0] = *(const v8h*)(&As[m * C0 + hh * 16]);
    fa0.h[1] = *(const v8h*)(&As[m * C0 + hh * 16 + 8]);
    fa1.h[0] = *(const v8h*)(&As[m * C0 + 32 + hh * 16]);
    fa1.h[1] = *(const v8h*)(&As[m * C0 + 40 + hh * 16]);
    v8f c = {};
    c = __builtin_amdgcn_wmma_f32_16x16x32_f16(false, fa0.v, false, fb0.v,
                                               (short)0, c, false, false);
    c = __builtin_amdgcn_wmma_f32_16x16x32_f16(false, fa1.v, false, fb1.v,
                                               (short)0, c, false, false);
    const float bias = b1[n0 + n];
#pragma unroll
    for (int v = 0; v < 8; ++v) {
        int mm = v + 8 * hh;
        h1h[(M0 + mm) * C1 + n0 + n] = (f16)(c[v] + bias);
    }
}

// --------------------------------------------- BN1 + ReLU + max over K
__global__ __launch_bounds__(128)
void maxpool_kernel(const f16* __restrict__ h1h, const float* __restrict__ bn1,
                    float* __restrict__ out) {
    const int bs = blockIdx.x, c = threadIdx.x;
    const float sc = bn1[c], sh = bn1[C1 + c];
    float mx = 0.0f;                    // relu output >= 0
    size_t base = (size_t)bs * NSAMP * C1 + c;
    for (int k = 0; k < NSAMP; ++k) {
        float v = fmaxf(sc * (float)h1h[base + (size_t)k * C1] + sh, 0.0f);
        mx = fmaxf(mx, v);
    }
    out[(size_t)NB * NPOINT * 3 + (size_t)bs * C1 + c] = mx;
}

extern "C" void kernel_launch(void* const* d_in, const int* in_sizes, int n_in,
                              void* d_out, int out_size, void* d_ws, size_t ws_size,
                              hipStream_t stream) {
    const float* xyz   = (const float*)d_in[0];
    const float* pts   = (const float*)d_in[1];
    const float* W0    = (const float*)d_in[2];
    const float* b0    = (const float*)d_in[3];
    const float* g0    = (const float*)d_in[4];
    const float* beta0 = (const float*)d_in[5];
    const float* W1    = (const float*)d_in[6];
    const float* b1    = (const float*)d_in[7];
    const float* g1    = (const float*)d_in[8];
    const float* beta1 = (const float*)d_in[9];
    float* out = (float*)d_out;

    // workspace layout
    char* w = (char*)d_ws;
    size_t off = 0;
    auto take = [&](size_t bytes) { void* p = w + off; off = (off + bytes + 255) & ~(size_t)255; return p; };
    int*   fps_idx = (int*)  take((size_t)NB * NPOINT * 4);
    float* bn0     = (float*)take(2 * C0 * 4);
    float* bn1     = (float*)take(2 * C1 * 4);
    float* part0   = (float*)take((size_t)512 * 2 * C0 * 4);
    float* part1   = (float*)take((size_t)512 * 2 * C1 * 4);
    f16*   W1h     = (f16*)  take((size_t)C1 * C0 * 2);
    f16*   h0h     = (f16*)  take((size_t)ROWS * C0 * 2);
    f16*   h1h     = (f16*)  take((size_t)ROWS * C1 * 2);
    (void)ws_size; (void)in_sizes; (void)n_in; (void)out_size;

    const float cnt = (float)ROWS;      // B*S*K samples per BN channel

    fps_kernel<<<NB, 1024, 0, stream>>>(xyz, fps_idx, out);
    convert_w1<<<(C1 * C0 + 255) / 256, 256, 0, stream>>>(W1, W1h);
    ballquery_gemm0<<<NB * NPOINT, 32, 0, stream>>>(xyz, pts, fps_idx, W0, b0, h0h);
    stats_partial<<<512, 256, 0, stream>>>(h0h, part0, (size_t)ROWS * C0, C0 - 1, C0);
    finalize_bn<<<1, C0, 0, stream>>>(part0, 512, C0, cnt, g0, beta0, bn0);
    gemm1_wmma<<<ROWS / 16, 256, 0, stream>>>(h0h, W1h, bn0, b1, h1h);
    stats_partial<<<512, 256, 0, stream>>>(h1h, part1, (size_t)ROWS * C1, C1 - 1, C1);
    finalize_bn<<<1, C1, 0, stream>>>(part1, 512, C1, cnt, g1, beta1, bn1);
    maxpool_kernel<<<NB * NPOINT, 128, 0, stream>>>(h1h, bn1, out);
}
